// Gemma4TextAttention_86320252715796
// MI455X (gfx1250) — matrix operand order repaired and checked by reading, not compiled, not run on hardware
//
#include <hip/hip_runtime.h>
#include <stdint.h>

// Problem constants (match reference)
#define kB   2
#define kS   2048
#define kHID 2048
#define kH   8
#define kHKV 4
#define kD   256
#define kWIN 512

typedef __attribute__((ext_vector_type(16))) _Float16 v16h;
typedef __attribute__((ext_vector_type(8)))  float    v8f;

#if defined(__gfx1250__) && __has_builtin(__builtin_amdgcn_tensor_load_to_lds)
#define USE_TDM 1
typedef unsigned int u32x4_t __attribute__((ext_vector_type(4)));
typedef int          i32x4_t __attribute__((ext_vector_type(4)));
typedef int          i32x8_t __attribute__((ext_vector_type(8)));
#else
#define USE_TDM 0
#endif

union H8 { uint4 u; _Float16 h[8]; };
union F2 { uint4 u[2]; v16h v; };

// ---- WMMA fragment loaders (wave32, layouts per CDNA5 ISA 7.12.2) ----
// A 16x32 f16: lane m=L&15, half=L>>4. VGPR0..3 = K[half*8 .. half*8+7],
// VGPR4..7 = K[16+half*8 .. 16+half*8+7]  -> two 16B loads per lane.
__device__ __forceinline__ v16h load_fragA(const _Float16* p, int ld) {
  int lane = threadIdx.x & 31;
  int m = lane & 15, hf = lane >> 4;
  F2 f;
  const char* base = (const char*)(p + (size_t)m * ld);
  f.u[0] = *(const uint4*)(base + hf * 16);
  f.u[1] = *(const uint4*)(base + 32 + hf * 16);
  return f.v;
}

// B 32x16 f16 staged column-contiguous ([n][k], stride ld): lane n=L&15, half=L>>4.
// VGPR0..3 = K[half*8 .. half*8+7], VGPR4..7 = K[16+half*8 ..]  -> two 16B loads per lane.
__device__ __forceinline__ v16h load_fragB(const _Float16* p, int ld) {
  int lane = threadIdx.x & 31;
  int n = lane & 15, hf = lane >> 4;
  F2 f;
  const char* base = (const char*)(p + (size_t)n * ld);
  f.u[0] = *(const uint4*)(base + hf * 16);
  f.u[1] = *(const uint4*)(base + 32 + hf * 16);
  return f.v;
}

__device__ __forceinline__ v8f wmma16(v16h a, v16h b, v8f c) {
  return __builtin_amdgcn_wmma_f32_16x16x32_f16(false, a, false, b, (short)0, c,
                                                false, false);
}

#if USE_TDM
// Issue a TDM 2D tile load: tile_h rows x tile_w elems (f16) from a row-major
// tensor with row length rowlen into LDS at ldsptr. D# per CDNA5 ISA.
__device__ __forceinline__ void tdm_load_2d(const void* gptr, void* ldsptr,
                                            int rowlen, int nrows, int tile_w,
                                            int tile_h) {
  unsigned long long ga = (unsigned long long)(uintptr_t)gptr;
  unsigned int lds = (unsigned int)(uintptr_t)ldsptr;
  // group0: count=1 | lds_addr | global_addr[56:0] | type=2 (bits 127:126)
  u32x4_t g0 = {1u, lds, (unsigned)(ga & 0xFFFFFFFFu),
                (unsigned)((ga >> 32) & 0x01FFFFFFu) | 0x80000000u};
  // group1: data_size=2B; tensor_dim0=rowlen, tensor_dim1=nrows;
  // tile_dim0=tile_w, tile_dim1=tile_h; tensor_dim0_stride=rowlen.
  i32x8_t g1 = {0x00010000,
                (int)((unsigned)(rowlen & 0xFFFF) << 16),
                (int)(((unsigned)rowlen >> 16) |
                      ((unsigned)(nrows & 0xFFFF) << 16)),
                (int)(((unsigned)nrows >> 16) | ((unsigned)tile_w << 16)),
                tile_h, rowlen, 0, 0};
  i32x4_t gz = {0, 0, 0, 0};
#if __clang_major__ >= 23
  i32x8_t g4 = {0, 0, 0, 0, 0, 0, 0, 0};
  __builtin_amdgcn_tensor_load_to_lds(g0, g1, gz, gz, g4, 0);
#else
  __builtin_amdgcn_tensor_load_to_lds(g0, g1, gz, gz, 0);
#endif
}
#endif

// ---------------------------------------------------------------------------
// f32 -> f16 conversion (grid-stride)
// ---------------------------------------------------------------------------
__global__ void cvt_f16_kernel(const float* __restrict__ in,
                               _Float16* __restrict__ out, int n) {
  for (int i = blockIdx.x * blockDim.x + threadIdx.x; i < n;
       i += gridDim.x * blockDim.x)
    out[i] = (_Float16)in[i];
}

// ---------------------------------------------------------------------------
// Tiled f16 WMMA GEMM: C[M,N] f32 = A[M,K] f16 (row-major) @ B[K,N] f16.
// 256 threads = 8 waves; 128x128 tile, K-step 32. Each wave: 64x32 (4x2 WMMA).
// Software-pipelined with double-buffered LDS: the Tensor Data Mover streams
// A tile kt+1 while WMMAs consume tile kt; B tile kt+1 rides a register
// pipeline. s_wait_tensorcnt is deferred until just before the barrier.
// ---------------------------------------------------------------------------
__global__ void __launch_bounds__(256)
gemm_f16_kernel(const _Float16* __restrict__ A, const _Float16* __restrict__ Bm,
                float* __restrict__ C, int M, int N, int K) {
  __shared__ __align__(16) _Float16 As[2][128 * 32];  // [m][k]
  __shared__ __align__(16) _Float16 Bs[2][128 * 32];  // [n][k] (transposed)
  const int tid  = threadIdx.x;
  const int wave = tid >> 5, lane = tid & 31;
  const int ncol = lane & 15, hf = lane >> 4;
  const int bm = blockIdx.y * 128, bn = blockIdx.x * 128;
  const int wm = (wave >> 2) * 64;   // wave row offset (0/64)
  const int wn = (wave & 3) * 32;    // wave col offset (0/32/64/96)
  v8f acc[4][2] = {};
  const int nk = K >> 5;

  uint4 breg[2];
#if !USE_TDM
  uint4 areg[2];
#endif

  // --- staging helpers -----------------------------------------------------
  auto issueA = [&](int kt, int buf) {
#if USE_TDM
    if (tid < 32)  // wave 0 drives the DMA; EXEC ignored by TDM
      tdm_load_2d(A + (size_t)bm * K + (size_t)kt * 32, &As[buf][0], K, M, 32,
                  128);
#else
    int r = tid >> 2, c = (tid & 3) * 8;
    areg[0] = *(const uint4*)&A[(size_t)(bm + r) * K + kt * 32 + c];
    areg[1] = *(const uint4*)&A[(size_t)(bm + r + 64) * K + kt * 32 + c];
    (void)buf;
#endif
  };
  auto commitA = [&](int buf) {
#if USE_TDM
    if (tid < 32) __builtin_amdgcn_s_wait_tensorcnt(0);
    (void)buf;
#else
    int r = tid >> 2, c = (tid & 3) * 8;
    *(uint4*)&As[buf][r * 32 + c] = areg[0];
    *(uint4*)&As[buf][(r + 64) * 32 + c] = areg[1];
#endif
  };
  auto issueB = [&](int kt) {
#pragma unroll
    for (int it = 0; it < 2; ++it) {
      int idx = tid + it * 256;
      int kk = idx >> 4, nn = (idx & 15) * 8;
      breg[it] = *(const uint4*)&Bm[(size_t)(kt * 32 + kk) * N + bn + nn];
    }
  };
  auto commitB = [&](int buf) {
#pragma unroll
    for (int it = 0; it < 2; ++it) {
      int idx = tid + it * 256;
      int kk = idx >> 4, nn = (idx & 15) * 8;
      H8 t;
      t.u = breg[it];
#pragma unroll
      for (int j = 0; j < 8; ++j) Bs[buf][(nn + j) * 32 + kk] = t.h[j];
    }
  };

  // --- prologue: stage tile 0 into buffer 0 --------------------------------
  issueA(0, 0);
  issueB(0);
  commitB(0);
  commitA(0);
  __syncthreads();

  // --- pipelined main loop -------------------------------------------------
  for (int kt = 0; kt < nk; ++kt) {
    const int cur = kt & 1, nxt = cur ^ 1;
    const bool more = (kt + 1) < nk;
    if (more) {  // start DMA + global loads for the next tile
      issueA(kt + 1, nxt);
      issueB(kt + 1);
    }
    if (kt + 2 < nk)  // global_prefetch_b8 for tile kt+2's B panel
      __builtin_prefetch(
          &Bm[(size_t)((kt + 2) * 32 + (tid >> 4)) * N + bn + (tid & 15) * 8],
          0, 1);

    // compute on the current buffer while the TDM streams the next A tile
    v16h af[4];
#pragma unroll
    for (int i = 0; i < 4; ++i)
      af[i] = load_fragA(&As[cur][(wm + i * 16) * 32], 32);
#pragma unroll
    for (int j = 0; j < 2; ++j) {
      v16h bf = load_fragB(&Bs[cur][(wn + j * 16) * 32], 32);
#pragma unroll
      for (int i = 0; i < 4; ++i) acc[i][j] = wmma16(af[i], bf, acc[i][j]);
    }

    if (more) {
      commitB(nxt);
      commitA(nxt);  // TDM: s_wait_tensorcnt 0 only now, after compute
    }
    __syncthreads();
  }

  // C/D layout: lane ncol=N, row = v + 8*half (CDNA5 ISA)
#pragma unroll
  for (int i = 0; i < 4; ++i)
#pragma unroll
    for (int j = 0; j < 2; ++j)
#pragma unroll
      for (int v = 0; v < 8; ++v) {
        int row = bm + wm + i * 16 + v + 8 * hf;
        int col = bn + wn + j * 16 + ncol;
        C[(size_t)row * N + col] = acc[i][j][v];
      }
}

// ---------------------------------------------------------------------------
// RMS-norm (+ optional weight, + optional RoPE), f32 in -> f16 out,
// output transposed to [B, nh, S, D]. One block = one (token, head), 256 thr.
// ---------------------------------------------------------------------------
__global__ void __launch_bounds__(256)
norm_rope_kernel(const float* __restrict__ X, const float* __restrict__ cosb,
                 const float* __restrict__ sinb, const float* __restrict__ w,
                 _Float16* __restrict__ out, int nh, int has_w, int use_rope) {
  const int token = blockIdx.x;   // b*S + s
  const int head  = blockIdx.y;
  const int d     = threadIdx.x;  // 0..255
  const int lane  = threadIdx.x & 31, wv = threadIdx.x >> 5;

  float x = X[((size_t)token * nh + head) * kD + d];
  float ss = x * x;
#pragma unroll
  for (int m = 16; m >= 1; m >>= 1) ss += __shfl_xor(ss, m, 32);
  __shared__ float red[8];
  __shared__ float sh[kD];
  if (lane == 0) red[wv] = ss;
  __syncthreads();
  float tot = 0.0f;
#pragma unroll
  for (int i = 0; i < 8; ++i) tot += red[i];
  float nv = x * rsqrtf(tot * (1.0f / (float)kD) + 1e-6f);
  if (has_w) nv *= (1.0f + w[d]);
  float val = nv;
  if (use_rope) {
    sh[d] = nv;
    __syncthreads();
    float rot = (d < kD / 2) ? -sh[d + kD / 2] : sh[d - kD / 2];
    float c = cosb[(size_t)token * kD + d];
    float s = sinb[(size_t)token * kD + d];
    val = nv * c + rot * s;
  }
  int b = token >> 11;       // / kS
  int s_ = token & (kS - 1);
  out[(((size_t)b * nh + head) * kS + s_) * kD + d] = (_Float16)val;
}

// ---------------------------------------------------------------------------
// Sliding-window flash attention with WMMA. Block = 64-query tile for one
// (b,h); 4 waves, each owns 16 queries; 32-key tiles with online softmax.
// K tile is streamed by the TDM (row-major 32x256 = a perfect 2D D# tile);
// the tensorcnt wait is deferred past the manual V-transpose staging.
// ---------------------------------------------------------------------------
__global__ void __launch_bounds__(128)
attn_kernel(const _Float16* __restrict__ Qh, const _Float16* __restrict__ Kh,
            const _Float16* __restrict__ Vh, _Float16* __restrict__ Oh) {
  __shared__ __align__(16) _Float16 Ks[32 * kD];     // [key][d]   16KB
  __shared__ __align__(16) _Float16 Vt[kD * 32];     // [d][key]   16KB
  __shared__ __align__(16) _Float16 Ps[4][16 * 32];  // per-wave P  4KB

  const int qt = blockIdx.x, h = blockIdx.y, b = blockIdx.z;
  const int hkv = h >> 1;  // groups = H/HKV = 2
  const int qbase = qt * 64;
  const int tid = threadIdx.x;
  const int wave = tid >> 5, lane = tid & 31;
  const int ncol = lane & 15, hf = lane >> 4;

  const _Float16* Qg = Qh + (((size_t)(b * kH + h) * kS) + qbase + wave * 16) * kD;
  const _Float16* Kg = Kh + ((size_t)(b * kHKV + hkv) * kS) * kD;
  const _Float16* Vg = Vh + ((size_t)(b * kHKV + hkv) * kS) * kD;

  v8f o[16] = {};
  float m_i[8], l_i[8];
#pragma unroll
  for (int v = 0; v < 8; ++v) { m_i[v] = -1e30f; l_i[v] = 0.0f; }

  int kstart = qbase - kWIN + 1;
  if (kstart < 0) kstart = 0;
  const int kt0 = kstart >> 5;
  const int kt1 = (qbase + 63) >> 5;

  for (int kt = kt0; kt <= kt1; ++kt) {
    const int k0 = kt * 32;
    __syncthreads();
#if USE_TDM
    if (tid < 32)  // DMA the K tile [32][256] while we transpose V below
      tdm_load_2d(Kg + (size_t)k0 * kD, &Ks[0], kD, kS, kD, 32);
#else
    for (int i = tid; i < 1024; i += 128) {
      int r = i >> 5, c = (i & 31) * 8;
      *(uint4*)&Ks[r * kD + c] = *(const uint4*)&Kg[(size_t)(k0 + r) * kD + c];
    }
#endif
    // stage V tile transposed -> [d][key]
    for (int i = tid; i < 1024; i += 128) {
      int r = i >> 5, c = (i & 31) * 8;
      H8 t;
      t.u = *(const uint4*)&Vg[(size_t)(k0 + r) * kD + c];
#pragma unroll
      for (int j = 0; j < 8; ++j) Vt[(c + j) * 32 + r] = t.h[j];
    }
#if USE_TDM
    if (tid < 32) __builtin_amdgcn_s_wait_tensorcnt(0);
#endif
    __syncthreads();

    // scores: 16 queries x 32 keys, K-dim = D = 256 in 8 WMMA steps
    v8f sc0 = {}, sc1 = {};
#pragma unroll
    for (int dk = 0; dk < 8; ++dk) {
      v16h qa  = load_fragA(Qg + dk * 32, kD);             // Q from global
      v16h kb0 = load_fragB(&Ks[dk * 32], kD);             // keys 0..15
      v16h kb1 = load_fragB(&Ks[16 * kD + dk * 32], kD);   // keys 16..31
      sc0 = wmma16(qa, kb0, sc0);
      sc1 = wmma16(qa, kb1, sc1);
    }

    // mask + online softmax (row r = v + 8*half, col = ncol)
    float alpha[8];
#pragma unroll
    for (int v = 0; v < 8; ++v) {
      int qpos = qbase + wave * 16 + v + 8 * hf;
      int d0 = qpos - (k0 + ncol);
      int d1 = qpos - (k0 + 16 + ncol);
      bool ok0 = (d0 >= 0) && (d0 < kWIN);
      bool ok1 = (d1 >= 0) && (d1 < kWIN);
      float s0 = ok0 ? sc0[v] : -1e30f;
      float s1 = ok1 ? sc1[v] : -1e30f;
      float rm = fmaxf(s0, s1);
#pragma unroll
      for (int msk = 8; msk >= 1; msk >>= 1)
        rm = fmaxf(rm, __shfl_xor(rm, msk, 32));
      float nm = fmaxf(m_i[v], rm);
      alpha[v] = __expf(m_i[v] - nm);
      float e0 = ok0 ? __expf(s0 - nm) : 0.0f;
      float e1 = ok1 ? __expf(s1 - nm) : 0.0f;
      float rs = e0 + e1;
#pragma unroll
      for (int msk = 8; msk >= 1; msk >>= 1) rs += __shfl_xor(rs, msk, 32);
      l_i[v] = l_i[v] * alpha[v] + rs;
      m_i[v] = nm;
      Ps[wave][(v + 8 * hf) * 32 + ncol]      = (_Float16)e0;
      Ps[wave][(v + 8 * hf) * 32 + 16 + ncol] = (_Float16)e1;
    }
#pragma unroll
    for (int t = 0; t < 16; ++t)
#pragma unroll
      for (int v = 0; v < 8; ++v) o[t][v] *= alpha[v];

    // PV: P(16x32) @ V(32x256) in 16 WMMA n-subtiles
    v16h pa = load_fragA(&Ps[wave][0], 32);
#pragma unroll
    for (int t = 0; t < 16; ++t) {
      v16h vb = load_fragB(&Vt[t * 16 * 32], 32);
      o[t] = wmma16(pa, vb, o[t]);
    }
  }

  // epilogue: normalize and write O as f16 [b*s][H*D]
#pragma unroll
  for (int t = 0; t < 16; ++t)
#pragma unroll
    for (int v = 0; v < 8; ++v) {
      int qpos = qbase + wave * 16 + v + 8 * hf;
      float val = o[t][v] / l_i[v];
      Oh[(size_t)(b * kS + qpos) * (kH * kD) + h * kD + t * 16 + ncol] =
          (_Float16)val;
    }
}

// ---------------------------------------------------------------------------
// Host orchestration
// ---------------------------------------------------------------------------
extern "C" void kernel_launch(void* const* d_in, const int* in_sizes, int n_in,
                              void* d_out, int out_size, void* d_ws,
                              size_t ws_size, hipStream_t stream) {
  (void)in_sizes; (void)n_in; (void)out_size; (void)ws_size;
  const float* hs   = (const float*)d_in[0];
  const float* cosb = (const float*)d_in[1];
  const float* sinb = (const float*)d_in[2];
  // d_in[3] = attention_mask: recomputed analytically (0 <= q-k < WINDOW)
  const float* Wq  = (const float*)d_in[4];
  const float* Wk  = (const float*)d_in[5];
  const float* Wv  = (const float*)d_in[6];
  const float* Wo  = (const float*)d_in[7];
  const float* qnw = (const float*)d_in[8];
  const float* knw = (const float*)d_in[9];
  float* out = (float*)d_out;

  char* ws = (char*)d_ws;
  size_t off = 0;
  auto alloc = [&](size_t bytes) -> void* {
    void* p = ws + off;
    off = (off + bytes + 255) & ~(size_t)255;
    return p;
  };
  const size_t M = (size_t)kB * kS;  // 4096 tokens
  _Float16* hsb = (_Float16*)alloc(M * kHID * 2);
  _Float16* Wqb = (_Float16*)alloc((size_t)kHID * kH * kD * 2);
  _Float16* Wkb = (_Float16*)alloc((size_t)kHID * kHKV * kD * 2);
  _Float16* Wvb = (_Float16*)alloc((size_t)kHID * kHKV * kD * 2);
  _Float16* Wob = (_Float16*)alloc((size_t)kH * kD * kHID * 2);
  float*    Qf  = (float*)alloc(M * kH * kD * 4);
  float*    Kf  = (float*)alloc(M * kHKV * kD * 4);
  float*    Vf  = (float*)alloc(M * kHKV * kD * 4);
  _Float16* Qh  = (_Float16*)alloc(M * kH * kD * 2);
  _Float16* Kh  = (_Float16*)alloc(M * kHKV * kD * 2);
  _Float16* Vh  = (_Float16*)alloc(M * kHKV * kD * 2);
  _Float16* Oh  = (_Float16*)alloc(M * kH * kD * 2);

  // 1) f32 -> f16
  cvt_f16_kernel<<<4096, 256, 0, stream>>>(hs, hsb, (int)(M * kHID));
  cvt_f16_kernel<<<4096, 256, 0, stream>>>(Wq, Wqb, kHID * kH * kD);
  cvt_f16_kernel<<<2048, 256, 0, stream>>>(Wk, Wkb, kHID * kHKV * kD);
  cvt_f16_kernel<<<2048, 256, 0, stream>>>(Wv, Wvb, kHID * kHKV * kD);
  cvt_f16_kernel<<<4096, 256, 0, stream>>>(Wo, Wob, kH * kD * kHID);

  // 2) QKV projections (f16 WMMA, f32 accumulate)
  gemm_f16_kernel<<<dim3(16, 32), 256, 0, stream>>>(hsb, Wqb, Qf, (int)M, 2048, 2048);
  gemm_f16_kernel<<<dim3(8, 32), 256, 0, stream>>>(hsb, Wkb, Kf, (int)M, 1024, 2048);
  gemm_f16_kernel<<<dim3(8, 32), 256, 0, stream>>>(hsb, Wvb, Vf, (int)M, 1024, 2048);

  // 3) RMS-norm (+RoPE for Q,K), transpose to [B,heads,S,D] f16
  norm_rope_kernel<<<dim3((int)M, kH), 256, 0, stream>>>(Qf, cosb, sinb, qnw, Qh, kH, 1, 1);
  norm_rope_kernel<<<dim3((int)M, kHKV), 256, 0, stream>>>(Kf, cosb, sinb, knw, Kh, kHKV, 1, 1);
  norm_rope_kernel<<<dim3((int)M, kHKV), 256, 0, stream>>>(Vf, cosb, sinb, nullptr, Vh, kHKV, 0, 0);

  // 4) sliding-window flash attention
  attn_kernel<<<dim3(kS / 64, kH, kB), 128, 0, stream>>>(Qh, Kh, Vh, Oh);

  // 5) output projection -> f32 d_out
  gemm_f16_kernel<<<dim3(16, 32), 256, 0, stream>>>(Oh, Wob, out, (int)M, 2048, 2048);
}
